// TransformerEncoderBlock_83511344103580
// MI455X (gfx1250) — matrix operand order repaired and checked by reading, not compiled, not run on hardware
//
#include <hip/hip_runtime.h>

// ---------------- problem constants ----------------
#define D_MODEL  1024
#define N_HEADS  16
#define HEAD_DIM 64
#define D_FF     4096
#define BATCH    2
#define SEQ      2048
#define M_ROWS   (BATCH * SEQ)   // 4096

// ---------------- vector types ----------------
typedef __attribute__((ext_vector_type(16))) __bf16 v16bf;
typedef __attribute__((ext_vector_type(8)))  __bf16 v8bf;
typedef __attribute__((ext_vector_type(8)))  float  v8f;
typedef __attribute__((ext_vector_type(4)))  int    v4i;

union AB16 { v16bf v; v8bf h[2]; };

// ---------------- CDNA5 async global->LDS (ASYNCcnt) ----------------
#if defined(__has_builtin)
#if __has_builtin(__builtin_amdgcn_global_load_async_to_lds_b128) && \
    __has_builtin(__builtin_amdgcn_s_wait_asynccnt)
#define HAVE_ASYNC_LDS 1
#endif
#endif
#ifndef HAVE_ASYNC_LDS
#define HAVE_ASYNC_LDS 0
#endif

typedef __attribute__((address_space(1))) v4i* gptr_v4i;
typedef __attribute__((address_space(3))) v4i* lptr_v4i;

// copy 16 bytes (8 x bf16) from global to LDS, per lane
static __device__ inline void async_copy16(const __bf16* g, __bf16* l) {
#if HAVE_ASYNC_LDS
  __builtin_amdgcn_global_load_async_to_lds_b128((gptr_v4i)g, (lptr_v4i)l, 0, 0);
#else
  *(v8bf*)l = *(const v8bf*)g;
#endif
}

static __device__ inline void async_wait() {
#if HAVE_ASYNC_LDS
  __builtin_amdgcn_s_wait_asynccnt(0);
#endif
}

// ---------------- DPP16 butterfly reductions (16-lane groups) --------------
// xor1 = quad_perm(1,0,3,2)=0xB1 ; xor2 = quad_perm(2,3,0,1)=0x4E ;
// row_half_mirror=0x141 acts as xor4 once quads are uniform ;
// row_mirror=0x140 acts as xor8 once half-rows are uniform.
template <int CTRL>
static __device__ inline float dpp_mov(float x) {
  return __builtin_bit_cast(
      float, __builtin_amdgcn_update_dpp(0, __builtin_bit_cast(int, x), CTRL,
                                         0xF, 0xF, true));
}
static __device__ inline float row16_max(float v) {
  v = fmaxf(v, dpp_mov<0xB1>(v));
  v = fmaxf(v, dpp_mov<0x4E>(v));
  v = fmaxf(v, dpp_mov<0x141>(v));
  v = fmaxf(v, dpp_mov<0x140>(v));
  return v;
}
static __device__ inline float row16_sum(float v) {
  v += dpp_mov<0xB1>(v);
  v += dpp_mov<0x4E>(v);
  v += dpp_mov<0x141>(v);
  v += dpp_mov<0x140>(v);
  return v;
}

// A-fragment (16xK tile, row-major, stride ldk elements):
// lane m = lane&15, half = lane>>4 ; element i -> K = (i<8 ? i : i+8) + 8*half
static __device__ inline v16bf load_a_frag(const __bf16* base, int ldk) {
  const int lane = threadIdx.x & 31;
  const int m    = lane & 15;
  const int half = lane >> 4;
  const __bf16* p = base + (size_t)m * ldk + 8 * half;
  AB16 r;
  r.h[0] = *(const v8bf*)(p);
  r.h[1] = *(const v8bf*)(p + 16);
  return r.v;
}

// B-fragment from an N-major (transposed) tile: base points at [n][k] storage,
// stride ldk ; lane n = lane&15, half = lane>>4 ; element i -> K = (i<8 ? i : i+8) + 8*half
static __device__ inline v16bf load_b_frag(const __bf16* base, int ldk) {
  const int lane = threadIdx.x & 31;
  const int n    = lane & 15;
  const int half = lane >> 4;
  const __bf16* p = base + (size_t)n * ldk + 8 * half;
  AB16 r;
  r.h[0] = *(const v8bf*)(p);
  r.h[1] = *(const v8bf*)(p + 16);
  return r.v;
}

static __device__ inline float gelu_exact(float x) {
  return 0.5f * x * (1.0f + erff(x * 0.70710678118654752f));
}

// ---------------- conversion kernels ----------------
__global__ __launch_bounds__(256) void cvt_f32_bf16(const float* __restrict__ in,
                                                    __bf16* __restrict__ out,
                                                    size_t n) {
  size_t i = (size_t)blockIdx.x * 256 + threadIdx.x;
  if (i < n) out[i] = (__bf16)in[i];
}

// out[n*K + k] = (bf16) in[k*N + n]   (weight [K,N] -> bf16 [N,K])
__global__ __launch_bounds__(256) void cvt_transpose_bf16(const float* __restrict__ in,
                                                          __bf16* __restrict__ out,
                                                          int K, int N) {
  int n = blockIdx.x * 64 + threadIdx.x;
  int k = blockIdx.y * 4 + threadIdx.y;
  if (n < N && k < K) out[(size_t)n * K + k] = (__bf16)in[(size_t)k * N + n];
}

// ---------------- generic WMMA GEMM (double-buffered async staging) ----------
// C[M,N] = A[M,K](bf16,row-major) @ W (given as Wt[N,K] bf16) + bias[N]
// Block: 128 threads (4 waves), tile BM=128, BN=64, BK=32.
// Each wave computes a 32x64 strip: 8 v_wmma per K-step.
template <int ACT, bool WF32, bool WBF16>
__global__ __launch_bounds__(128)
void gemm_bias(const __bf16* __restrict__ A, const __bf16* __restrict__ Wt,
               const float* __restrict__ bias, float* __restrict__ Cf,
               __bf16* __restrict__ Cb, int M, int N, int K) {
  __shared__ __align__(16) __bf16 As[2][128 * 32];
  __shared__ __align__(16) __bf16 Bs[2][64 * 32];

  const int tid  = threadIdx.x;
  const int wave = tid >> 5;
  const int lane = tid & 31;
  const int l15  = lane & 15;
  const int half = lane >> 4;
  const int m0   = blockIdx.y * 128;
  const int n0   = blockIdx.x * 64;

  v8f acc[2][4];
#pragma unroll
  for (int rt = 0; rt < 2; ++rt)
#pragma unroll
    for (int i = 0; i < 4; ++i)
#pragma unroll
      for (int j = 0; j < 8; ++j) acc[rt][i][j] = 0.0f;

  // stage one BK=32 tile (A: 128x32, Wt: 64x32) into LDS buffer `buf`
  auto stage = [&](int buf, int k0) {
#pragma unroll
    for (int j = 0; j < 4; ++j) {
      int v   = tid + 128 * j;
      int row = v >> 2;          // 4 x v8bf per 32-wide row
      int co  = (v & 3) * 8;
      async_copy16(&A[(size_t)(m0 + row) * K + k0 + co], &As[buf][row * 32 + co]);
    }
#pragma unroll
    for (int j = 0; j < 2; ++j) {
      int v   = tid + 128 * j;
      int row = v >> 2;
      int co  = (v & 3) * 8;
      async_copy16(&Wt[(size_t)(n0 + row) * K + k0 + co], &Bs[buf][row * 32 + co]);
    }
  };

  stage(0, 0);
  async_wait();
  __syncthreads();

  int buf = 0;
  for (int k0 = 0; k0 < K; k0 += 32) {
    if (k0 + 32 < K) stage(buf ^ 1, k0 + 32);   // overlap with WMMA below

    v16bf a0 = load_a_frag(&As[buf][(wave * 32) * 32], 32);
    v16bf a1 = load_a_frag(&As[buf][(wave * 32 + 16) * 32], 32);
#pragma unroll
    for (int nt = 0; nt < 4; ++nt) {
      v16bf bfr = load_b_frag(&Bs[buf][(nt * 16) * 32], 32);
      acc[0][nt] = __builtin_amdgcn_wmma_f32_16x16x32_bf16(
          false, a0, false, bfr, (short)0, acc[0][nt], false, false);
      acc[1][nt] = __builtin_amdgcn_wmma_f32_16x16x32_bf16(
          false, a1, false, bfr, (short)0, acc[1][nt], false, false);
    }
    async_wait();
    __syncthreads();
    buf ^= 1;
  }

#pragma unroll
  for (int rt = 0; rt < 2; ++rt)
#pragma unroll
    for (int nt = 0; nt < 4; ++nt)
#pragma unroll
      for (int r = 0; r < 8; ++r) {
        int row = m0 + wave * 32 + rt * 16 + r + 8 * half;
        int col = n0 + nt * 16 + l15;
        float v = acc[rt][nt][r] + bias[col];
        if (ACT == 1) v = gelu_exact(v);
        if (WF32)  Cf[(size_t)row * N + col] = v;
        if (WBF16) Cb[(size_t)row * N + col] = (__bf16)v;
      }
}

// ---------------- flash attention (bf16 WMMA) ----------------
// Q,K,V,Ctx are bf16 [B,S,D_MODEL]; head h occupies columns [h*64, h*64+64).
__global__ __launch_bounds__(128)
void attention(const __bf16* __restrict__ Q, const __bf16* __restrict__ Kb,
               const __bf16* __restrict__ Vb, __bf16* __restrict__ Ctx) {
  __shared__ __align__(16) __bf16 Ks[32 * 64];      // [key][hd]
  __shared__ __align__(16) __bf16 Vt[64 * 32];      // [hd][key] (transposed)
  __shared__ __align__(16) __bf16 Ps[4][16 * 32];   // per-wave P tile [q][key]

  const int tid  = threadIdx.x;
  const int wave = tid >> 5;
  const int lane = tid & 31;
  const int l15  = lane & 15;
  const int half = lane >> 4;

  const int bh = blockIdx.y;
  const int b  = bh / N_HEADS;
  const int h  = bh % N_HEADS;
  const int q0 = blockIdx.x * 64 + wave * 16;
  const size_t baseBS = (size_t)b * SEQ;
  const size_t hoff   = (size_t)h * HEAD_DIM;

  // Q tile: two A-frags (hd chunks 0..31 and 32..63)
  AB16 aq[2];
#pragma unroll
  for (int c = 0; c < 2; ++c) {
    const __bf16* p = Q + (baseBS + q0 + l15) * D_MODEL + hoff + 32 * c + 8 * half;
    aq[c].h[0] = *(const v8bf*)(p);
    aq[c].h[1] = *(const v8bf*)(p + 16);
  }

  float mrow[8], lrow[8];
  v8f acc[4];
#pragma unroll
  for (int r = 0; r < 8; ++r) { mrow[r] = -1e30f; lrow[r] = 0.0f; }
#pragma unroll
  for (int i = 0; i < 4; ++i)
#pragma unroll
    for (int j = 0; j < 8; ++j) acc[i][j] = 0.0f;

  const float scale = 0.125f;   // 1/sqrt(64)

  for (int kv0 = 0; kv0 < SEQ; kv0 += 32) {
    // cooperative stage: K row-major via async-to-LDS, V transposed via VALU
#pragma unroll
    for (int j = 0; j < 2; ++j) {
      int v   = tid + 128 * j;
      int row = v >> 3;          // 8 x v8bf per 64-wide row
      int co  = (v & 7) * 8;
      async_copy16(&Kb[(baseBS + kv0 + row) * D_MODEL + hoff + co],
                   &Ks[row * 64 + co]);
      v8bf vv = *(const v8bf*)&Vb[(baseBS + kv0 + row) * D_MODEL + hoff + co];
#pragma unroll
      for (int e = 0; e < 8; ++e) Vt[(co + e) * 32 + row] = vv[e];
      if (kv0 + 32 < SEQ) {      // CDNA5 global_prefetch_b8 of next V chunk
        __builtin_prefetch(&Vb[(baseBS + kv0 + 32 + row) * D_MODEL + hoff + co], 0, 0);
      }
    }
    async_wait();
    __syncthreads();

    // scores: two 16x16 tiles (keys kv0..+15, kv0+16..+31)
    v8f s[2];
#pragma unroll
    for (int t = 0; t < 2; ++t) {
#pragma unroll
      for (int j = 0; j < 8; ++j) s[t][j] = 0.0f;
#pragma unroll
      for (int c = 0; c < 2; ++c) {
        v16bf bk = load_b_frag(&Ks[(t * 16) * 64 + 32 * c], 64);
        s[t] = __builtin_amdgcn_wmma_f32_16x16x32_bf16(
            false, aq[c].v, false, bk, (short)0, s[t], false, false);
      }
#pragma unroll
      for (int j = 0; j < 8; ++j) s[t][j] *= scale;
    }

    // online softmax stats (rows live in 16-lane groups, DPP16 butterflies)
    float mnew[8];
#pragma unroll
    for (int r = 0; r < 8; ++r) {
      float v = row16_max(fmaxf(s[0][r], s[1][r]));
      mnew[r] = fmaxf(mrow[r], v);
    }
#pragma unroll
    for (int t = 0; t < 2; ++t)
#pragma unroll
      for (int r = 0; r < 8; ++r) {
        float p = __expf(s[t][r] - mnew[r]);
        s[t][r] = p;
        Ps[wave][(r + 8 * half) * 32 + 16 * t + l15] = (__bf16)p;
      }
#pragma unroll
    for (int r = 0; r < 8; ++r) {
      float v = row16_sum(s[0][r] + s[1][r]);
      float f = __expf(mrow[r] - mnew[r]);
      lrow[r] = lrow[r] * f + v;
      mrow[r] = mnew[r];
#pragma unroll
      for (int nt = 0; nt < 4; ++nt) acc[nt][r] *= f;
    }

    // P (C-layout) -> A-layout via per-wave LDS round trip
    AB16 ap;
    const __bf16* pp = &Ps[wave][0];
    ap.h[0] = *(const v8bf*)&pp[l15 * 32 + 8 * half];
    ap.h[1] = *(const v8bf*)&pp[l15 * 32 + 16 + 8 * half];

#pragma unroll
    for (int nt = 0; nt < 4; ++nt) {
      v16bf bv = load_b_frag(&Vt[(16 * nt) * 32], 32);
      acc[nt] = __builtin_amdgcn_wmma_f32_16x16x32_bf16(
          false, ap.v, false, bv, (short)0, acc[nt], false, false);
    }
    __syncthreads();
  }

#pragma unroll
  for (int nt = 0; nt < 4; ++nt)
#pragma unroll
    for (int r = 0; r < 8; ++r) {
      int q   = q0 + r + 8 * half;
      int col = nt * 16 + l15;
      Ctx[(baseBS + q) * D_MODEL + hoff + col] = (__bf16)(acc[nt][r] / lrow[r]);
    }
}

// ---------------- residual add + LayerNorm ----------------
__global__ __launch_bounds__(256)
void add_ln(const float* __restrict__ X, const float* __restrict__ Y,
            const float* __restrict__ g, const float* __restrict__ be,
            float* __restrict__ Of, __bf16* __restrict__ Ob) {
  const int row = blockIdx.x;
  const float* x = X + (size_t)row * D_MODEL;
  const float* y = Y + (size_t)row * D_MODEL;
  __shared__ float red[256];
  const int tid = threadIdx.x;

  float s = 0.0f;
  for (int i = tid; i < D_MODEL; i += 256) s += x[i] + y[i];
  red[tid] = s;
  __syncthreads();
  for (int off = 128; off; off >>= 1) {
    if (tid < off) red[tid] += red[tid + off];
    __syncthreads();
  }
  const float mu = red[0] * (1.0f / D_MODEL);
  __syncthreads();

  float v = 0.0f;
  for (int i = tid; i < D_MODEL; i += 256) {
    float d = x[i] + y[i] - mu;
    v += d * d;
  }
  red[tid] = v;
  __syncthreads();
  for (int off = 128; off; off >>= 1) {
    if (tid < off) red[tid] += red[tid + off];
    __syncthreads();
  }
  const float rstd = rsqrtf(red[0] * (1.0f / D_MODEL) + 1e-5f);

  for (int i = tid; i < D_MODEL; i += 256) {
    float o = (x[i] + y[i] - mu) * rstd * g[i] + be[i];
    Of[(size_t)row * D_MODEL + i] = o;
    if (Ob) Ob[(size_t)row * D_MODEL + i] = (__bf16)o;
  }
}

// ---------------- host orchestration ----------------
extern "C" void kernel_launch(void* const* d_in, const int* in_sizes, int n_in,
                              void* d_out, int out_size, void* d_ws, size_t ws_size,
                              hipStream_t stream) {
  (void)in_sizes; (void)n_in; (void)out_size; (void)ws_size;

  const float* x   = (const float*)d_in[0];
  const float* Wq  = (const float*)d_in[1];
  const float* bq  = (const float*)d_in[2];
  const float* Wk  = (const float*)d_in[3];
  const float* bk  = (const float*)d_in[4];
  const float* Wv  = (const float*)d_in[5];
  const float* bv  = (const float*)d_in[6];
  const float* Wo  = (const float*)d_in[7];
  const float* bo  = (const float*)d_in[8];
  const float* g1  = (const float*)d_in[9];
  const float* b1  = (const float*)d_in[10];
  const float* g2  = (const float*)d_in[11];
  const float* b2  = (const float*)d_in[12];
  const float* W1  = (const float*)d_in[13];
  const float* bf1 = (const float*)d_in[14];
  const float* W2  = (const float*)d_in[15];
  const float* bf2 = (const float*)d_in[16];

  size_t off = 0;
  auto carve = [&](size_t bytes) -> char* {
    char* p = (char*)d_ws + off;
    off += (bytes + 255) & ~(size_t)255;
    return p;
  };
  __bf16* xb   = (__bf16*)carve((size_t)M_ROWS * D_MODEL * 2);
  __bf16* wqb  = (__bf16*)carve((size_t)D_MODEL * D_MODEL * 2);
  __bf16* wkb  = (__bf16*)carve((size_t)D_MODEL * D_MODEL * 2);
  __bf16* wvb  = (__bf16*)carve((size_t)D_MODEL * D_MODEL * 2);
  __bf16* wob  = (__bf16*)carve((size_t)D_MODEL * D_MODEL * 2);
  __bf16* w1b  = (__bf16*)carve((size_t)D_MODEL * D_FF * 2);
  __bf16* w2b  = (__bf16*)carve((size_t)D_FF * D_MODEL * 2);
  __bf16* qb   = (__bf16*)carve((size_t)M_ROWS * D_MODEL * 2);
  __bf16* kb   = (__bf16*)carve((size_t)M_ROWS * D_MODEL * 2);
  __bf16* vb   = (__bf16*)carve((size_t)M_ROWS * D_MODEL * 2);
  __bf16* ctxb = (__bf16*)carve((size_t)M_ROWS * D_MODEL * 2);
  float*  attf = (float*)carve((size_t)M_ROWS * D_MODEL * 4);
  float*  hf   = (float*)carve((size_t)M_ROWS * D_MODEL * 4);
  __bf16* hb   = (__bf16*)carve((size_t)M_ROWS * D_MODEL * 2);
  __bf16* ub   = (__bf16*)carve((size_t)M_ROWS * D_FF * 2);
  float*  f2   = (float*)carve((size_t)M_ROWS * D_MODEL * 4);

  // 1) x -> bf16
  {
    size_t n = (size_t)M_ROWS * D_MODEL;
    cvt_f32_bf16<<<dim3((unsigned)((n + 255) / 256)), dim3(256), 0, stream>>>(x, xb, n);
  }
  // 2) weights -> bf16, transposed [N,K]
  {
    dim3 blk(64, 4);
    cvt_transpose_bf16<<<dim3(D_MODEL / 64, D_MODEL / 4), blk, 0, stream>>>(Wq, wqb, D_MODEL, D_MODEL);
    cvt_transpose_bf16<<<dim3(D_MODEL / 64, D_MODEL / 4), blk, 0, stream>>>(Wk, wkb, D_MODEL, D_MODEL);
    cvt_transpose_bf16<<<dim3(D_MODEL / 64, D_MODEL / 4), blk, 0, stream>>>(Wv, wvb, D_MODEL, D_MODEL);
    cvt_transpose_bf16<<<dim3(D_MODEL / 64, D_MODEL / 4), blk, 0, stream>>>(Wo, wob, D_MODEL, D_MODEL);
    cvt_transpose_bf16<<<dim3(D_FF / 64, D_MODEL / 4), blk, 0, stream>>>(W1, w1b, D_MODEL, D_FF);
    cvt_transpose_bf16<<<dim3(D_MODEL / 64, D_FF / 4), blk, 0, stream>>>(W2, w2b, D_FF, D_MODEL);
  }
  // 3) Q/K/V projections (bf16 out)
  {
    dim3 grid(D_MODEL / 64, M_ROWS / 128), blk(128);
    gemm_bias<0, false, true><<<grid, blk, 0, stream>>>(xb, wqb, bq, nullptr, qb, M_ROWS, D_MODEL, D_MODEL);
    gemm_bias<0, false, true><<<grid, blk, 0, stream>>>(xb, wkb, bk, nullptr, kb, M_ROWS, D_MODEL, D_MODEL);
    gemm_bias<0, false, true><<<grid, blk, 0, stream>>>(xb, wvb, bv, nullptr, vb, M_ROWS, D_MODEL, D_MODEL);
  }
  // 4) flash attention
  attention<<<dim3(SEQ / 64, BATCH * N_HEADS), dim3(128), 0, stream>>>(qb, kb, vb, ctxb);
  // 5) output projection (f32 out)
  gemm_bias<0, true, false><<<dim3(D_MODEL / 64, M_ROWS / 128), dim3(128), 0, stream>>>(
      ctxb, wob, bo, attf, nullptr, M_ROWS, D_MODEL, D_MODEL);
  // 6) h = LN(x + attn_out)  (f32 + bf16)
  add_ln<<<dim3(M_ROWS), dim3(256), 0, stream>>>(x, attf, g1, b1, hf, hb);
  // 7) FFN1 + exact GELU (bf16 out)
  gemm_bias<1, false, true><<<dim3(D_FF / 64, M_ROWS / 128), dim3(128), 0, stream>>>(
      hb, w1b, bf1, nullptr, ub, M_ROWS, D_FF, D_MODEL);
  // 8) FFN2 (f32 out)
  gemm_bias<0, true, false><<<dim3(D_MODEL / 64, M_ROWS / 128), dim3(128), 0, stream>>>(
      ub, w2b, bf2, f2, nullptr, M_ROWS, D_MODEL, D_FF);
  // 9) out = LN(h + ffn)
  add_ln<<<dim3(M_ROWS), dim3(256), 0, stream>>>(hf, f2, g2, b2, (float*)d_out, nullptr);
}
